// RowGNNLayer_78778290143557
// MI455X (gfx1250) — matrix operand order repaired and checked by reading, not compiled, not run on hardware
//
#include <hip/hip_runtime.h>

#define N_NODES 50000
#define N_EDGES 600000
#define DIM     128

typedef __attribute__((ext_vector_type(4)))  _Float16 half4_t;
typedef __attribute__((ext_vector_type(8)))  _Float16 half8_t;
typedef __attribute__((ext_vector_type(16))) _Float16 v16h;
typedef __attribute__((ext_vector_type(8)))  float    v8f;

// ---------------------------------------------------------------------------
// Kernel 1: zero neighbor accumulator + degree (contiguous in workspace)
// ---------------------------------------------------------------------------
__global__ void k_zero(float4* __restrict__ p, int n4) {
    int i = blockIdx.x * blockDim.x + threadIdx.x;
    if (i < n4) {
        float4 z; z.x = 0.f; z.y = 0.f; z.z = 0.f; z.w = 0.f;
        p[i] = z;
    }
}

// ---------------------------------------------------------------------------
// Kernel 2: convert both weight matrices to f16 (row-major, same layout as
// input) and fuse the two bias vectors.
// ---------------------------------------------------------------------------
__global__ void k_prep(const float* __restrict__ Ws, const float* __restrict__ Wn,
                       const float* __restrict__ bs, const float* __restrict__ bn,
                       _Float16* __restrict__ Whs, _Float16* __restrict__ Whn,
                       float* __restrict__ bias) {
    int i = blockIdx.x * blockDim.x + threadIdx.x;
    if (i < DIM * DIM) {
        Whs[i] = (_Float16)Ws[i];
        Whn[i] = (_Float16)Wn[i];
    }
    if (i < DIM) bias[i] = bs[i] + bn[i];
}

// ---------------------------------------------------------------------------
// Kernel 3: edge scatter. One wave32 per edge; lane l handles 4 floats via
// float4 gather from x[src] and 4x GLOBAL_ATOMIC_ADD_F32 into neighbor[dst].
// x and neighbor are L2-resident (25.6 MB each vs 192 MB L2), so the atomics
// resolve in the L2 atomic units without touching HBM.
// ---------------------------------------------------------------------------
__global__ void k_scatter(const float* __restrict__ x, const int* __restrict__ ei,
                          const float* __restrict__ ew, float* __restrict__ neigh,
                          float* __restrict__ deg) {
    int t = blockIdx.x * blockDim.x + threadIdx.x;
    int e = t >> 5;
    int lane = t & 31;
    if (e >= N_EDGES) return;
    int src = ei[e];
    int dst = ei[N_EDGES + e];
    float w = ew[e];
    float4 v = ((const float4*)(x + (size_t)src * DIM))[lane];
    float* nb = neigh + (size_t)dst * DIM + lane * 4;
    __hip_atomic_fetch_add(nb + 0, v.x * w, __ATOMIC_RELAXED, __HIP_MEMORY_SCOPE_AGENT);
    __hip_atomic_fetch_add(nb + 1, v.y * w, __ATOMIC_RELAXED, __HIP_MEMORY_SCOPE_AGENT);
    __hip_atomic_fetch_add(nb + 2, v.z * w, __ATOMIC_RELAXED, __HIP_MEMORY_SCOPE_AGENT);
    __hip_atomic_fetch_add(nb + 3, v.w * w, __ATOMIC_RELAXED, __HIP_MEMORY_SCOPE_AGENT);
    if (lane == 0)
        __hip_atomic_fetch_add(deg + dst, w, __ATOMIC_RELAXED, __HIP_MEMORY_SCOPE_AGENT);
}

// ---------------------------------------------------------------------------
// Kernel 4: fused dual GEMM on WMMA.
//   out[m,n] = bias[n] + sum_k xh[m,k]*Whs[n,k] + sum_k nh[m,k]*Whn[n,k]
// Block = 256 threads = 8 waves; 64 rows per block staged to LDS as f16
// (neighbor normalized by clip(degree,1) during staging). Wave j owns output
// columns [16j,16j+16); B fragments (W.T tiles) live in VGPRs for all
// 4 k-steps x 2 matrices. Per M-tile all 8 A fragments are preloaded from
// LDS first, then 8 v_wmma_f32_16x16x32_f16 issue back-to-back (XDL co-exec).
// Stores: block-uniform fast path (8 immediate-offset global_store_b32),
// guarded path only in the final partial block.
// ---------------------------------------------------------------------------
__global__ __launch_bounds__(256) void k_gemm(
    const float* __restrict__ x, const float* __restrict__ neigh,
    const float* __restrict__ deg, const _Float16* __restrict__ Whs,
    const _Float16* __restrict__ Whn, const float* __restrict__ bias,
    float* __restrict__ out) {
    __shared__ _Float16 sx[64][136];  // +8 halves padding per row
    __shared__ _Float16 sn[64][136];

    const int tid  = threadIdx.x;
    const int row0 = blockIdx.x * 64;
    const bool full = (row0 + 64) <= N_NODES;   // block-uniform

    // Stage A operands (x and normalized neighbor) to LDS as f16.
    for (int i = tid; i < 64 * 32; i += 256) {
        int r  = i >> 5;        // local row
        int c4 = i & 31;        // float4 chunk within row
        int row = row0 + r;
        if (row > N_NODES - 1) row = N_NODES - 1;   // clamp; stores guarded later
        float4 vx = ((const float4*)x)[(size_t)row * 32 + c4];
        float4 vn = ((const float4*)neigh)[(size_t)row * 32 + c4];
        float invd = 1.0f / fmaxf(deg[row], 1.0f);
        half4_t hx = { (_Float16)vx.x, (_Float16)vx.y, (_Float16)vx.z, (_Float16)vx.w };
        half4_t hn = { (_Float16)(vn.x * invd), (_Float16)(vn.y * invd),
                       (_Float16)(vn.z * invd), (_Float16)(vn.w * invd) };
        *(half4_t*)&sx[r][c4 * 4] = hx;
        *(half4_t*)&sn[r][c4 * 4] = hn;
    }

    const int wave = tid >> 5;            // column tile j = wave
    const int lane = tid & 31;
    const int n    = (wave << 4) + (lane & 15);   // output column
    const int koff = (lane & 16) ? 16 : 0;        // B: lanes>=16 hold K+16..K+31

    // Preload B fragments: B[k, n] = W[n, k] -> contiguous 16-half runs of W row n.
    v16h bfs[4], bfn[4];
#pragma unroll
    for (int s = 0; s < 4; ++s) {
        const _Float16* ps = Whs + (size_t)n * DIM + 32 * s + koff;
        const _Float16* pn = Whn + (size_t)n * DIM + 32 * s + koff;
        half8_t lo = *(const half8_t*)ps;
        half8_t hi = *(const half8_t*)(ps + 8);
        bfs[s] = __builtin_shufflevector(lo, hi, 0,1,2,3,4,5,6,7,8,9,10,11,12,13,14,15);
        lo = *(const half8_t*)pn;
        hi = *(const half8_t*)(pn + 8);
        bfn[s] = __builtin_shufflevector(lo, hi, 0,1,2,3,4,5,6,7,8,9,10,11,12,13,14,15);
    }
    const float bsum = bias[n];

    __syncthreads();

#pragma unroll
    for (int mt = 0; mt < 4; ++mt) {
        const int ar = (mt << 4) + (lane & 15);     // A row (local)
        const int ab = (lane & 16) ? 16 : 0;        // A: lanes>=16 hold K+16..K+31

        // Preload all 8 A fragments for this M-tile, then issue WMMAs densely.
        v16h ax[4], an[4];
#pragma unroll
        for (int s = 0; s < 4; ++s) {
            half8_t lo = *(const half8_t*)&sx[ar][32 * s + ab];
            half8_t hi = *(const half8_t*)&sx[ar][32 * s + ab + 8];
            ax[s] = __builtin_shufflevector(lo, hi, 0,1,2,3,4,5,6,7,8,9,10,11,12,13,14,15);
            lo = *(const half8_t*)&sn[ar][32 * s + ab];
            hi = *(const half8_t*)&sn[ar][32 * s + ab + 8];
            an[s] = __builtin_shufflevector(lo, hi, 0,1,2,3,4,5,6,7,8,9,10,11,12,13,14,15);
        }

        v8f acc = { bsum, bsum, bsum, bsum, bsum, bsum, bsum, bsum };
#pragma unroll
        for (int s = 0; s < 4; ++s) {
            acc = __builtin_amdgcn_wmma_f32_16x16x32_f16(false, ax[s], false, bfs[s],
                                                         (short)0, acc, false, false);
            acc = __builtin_amdgcn_wmma_f32_16x16x32_f16(false, an[s], false, bfn[s],
                                                         (short)0, acc, false, false);
        }

        // C/D layout: VGPR r, lane L -> (M = r + 8*(L>>4), N = L&15)
        const int mrow = row0 + (mt << 4) + ((lane & 16) ? 8 : 0);
        float* p = out + (size_t)mrow * DIM + n;    // stride DIM floats per reg
        if (full) {
#pragma unroll
            for (int r2 = 0; r2 < 8; ++r2)
                p[(size_t)r2 * DIM] = acc[r2];      // immediate-offset b32 stores
        } else {
#pragma unroll
            for (int r2 = 0; r2 < 8; ++r2)
                if (mrow + r2 < N_NODES) p[(size_t)r2 * DIM] = acc[r2];
        }
    }
}

// ---------------------------------------------------------------------------
extern "C" void kernel_launch(void* const* d_in, const int* in_sizes, int n_in,
                              void* d_out, int out_size, void* d_ws, size_t ws_size,
                              hipStream_t stream) {
    const float* x      = (const float*)d_in[0];
    const int*   ei     = (const int*)d_in[1];
    const float* ew     = (const float*)d_in[2];
    const float* W_self = (const float*)d_in[3];
    const float* b_self = (const float*)d_in[4];
    const float* W_neig = (const float*)d_in[5];
    const float* b_neig = (const float*)d_in[6];
    float* out = (float*)d_out;

    // Workspace layout (bytes):
    //   [0, 25600000)            neighbor f32 [50000][128]
    //   [25600000, 25800000)     degree   f32 [50000]
    //   [25800000, 25832768)     W_self   f16 [128][128]
    //   [25832768, 25865536)     W_neigh  f16 [128][128]
    //   [25865536, 25866048)     bias     f32 [128]
    char* ws = (char*)d_ws;
    float*    neigh = (float*)ws;
    float*    deg   = (float*)(ws + 25600000);
    _Float16* Whs   = (_Float16*)(ws + 25800000);
    _Float16* Whn   = (_Float16*)(ws + 25832768);
    float*    bias  = (float*)(ws + 25865536);

    // 1. zero neighbor + degree (contiguous: 6,450,000 floats = 1,612,500 float4)
    {
        int n4 = (N_NODES * DIM + N_NODES) / 4;
        int blocks = (n4 + 255) / 256;
        k_zero<<<blocks, 256, 0, stream>>>((float4*)ws, n4);
    }
    // 2. weight f16 conversion + bias fusion
    k_prep<<<(DIM * DIM + 255) / 256, 256, 0, stream>>>(W_self, W_neig, b_self, b_neig,
                                                        Whs, Whn, bias);
    // 3. edge scatter: one wave32 per edge
    {
        long long threads = (long long)N_EDGES * 32;
        int blocks = (int)((threads + 255) / 256);
        k_scatter<<<blocks, 256, 0, stream>>>(x, ei, ew, neigh, deg);
    }
    // 4. fused dual WMMA GEMM, 64 rows per block
    {
        int blocks = (N_NODES + 63) / 64;
        k_gemm<<<blocks, 256, 0, stream>>>(x, neigh, deg, Whs, Whn, bias, out);
    }
}
